// Attention_24781961298510
// MI455X (gfx1250) — matrix operand order repaired and checked by reading, not compiled, not run on hardware
//
#include <hip/hip_runtime.h>
#include <stdint.h>

// ---------------------------------------------------------------------------
// Types matching the CDNA5 WMMA builtin signatures (probe-confirmed):
//   __builtin_amdgcn_wmma_f32_16x16x32_bf16(bool, v16bf, bool, v16bf,
//                                           short, v8f, bool, bool)
// ---------------------------------------------------------------------------
typedef __bf16 bf16;
typedef bf16  v8bf  __attribute__((ext_vector_type(8)));
typedef bf16  v16bf __attribute__((ext_vector_type(16)));
typedef float v8f   __attribute__((ext_vector_type(8)));

#define EMB   1024
#define HEADS 16
#define HD    64
#define BATCH 4
#define SEQ   2048
#define ROWS  (BATCH * SEQ)        // 8192
#define KSTEPS (EMB / 32)          // 32

// Pack two contiguous v8bf halves into the v16bf A/B fragment register shape.
static __device__ __forceinline__ v16bf pack16(const bf16* lo, const bf16* hi) {
  v8bf l = *(const v8bf*)lo;
  v8bf h = *(const v8bf*)hi;
  v16bf r;
#pragma unroll
  for (int i = 0; i < 8; ++i) { r[i] = l[i]; r[i + 8] = h[i]; }
  return r;
}

static __device__ __forceinline__ v8f wmma_bf16(v16bf a, v16bf b, v8f c) {
  return __builtin_amdgcn_wmma_f32_16x16x32_bf16(
      false, a, false, b, (short)0, c, false, false);
}

// Async global -> LDS copy (CDNA5 GLOBAL_LOAD_ASYNC_TO_LDS_B128, GV mode:
// VDST = LDS byte offset VGPR, VADDR = 64-bit global address, ASYNCcnt).
static __device__ __forceinline__ void async_b128_x2(uint32_t lds0,
                                                     uint32_t lds1,
                                                     const bf16* src0,
                                                     const bf16* src1) {
  asm volatile(
      "global_load_async_to_lds_b128 %0, %2, off\n\t"
      "global_load_async_to_lds_b128 %1, %3, off"
      :: "v"(lds0), "v"(lds1), "v"(src0), "v"(src1)
      : "memory");
}

static __device__ __forceinline__ void wait_async0() {
  asm volatile("s_wait_asynccnt 0x0" ::: "memory");
}

// ---------------------------------------------------------------------------
// Prep kernels: fp32 -> bf16 (and transposed weights so every WMMA B-operand
// load is a pair of contiguous 16B runs in K).
// ---------------------------------------------------------------------------
__global__ void cvt_bf16_kernel(const float* __restrict__ in,
                                bf16* __restrict__ out, int n) {
  int i = blockIdx.x * 256 + threadIdx.x;
  if (i < n) out[i] = (bf16)in[i];
}

// out[n*K + k] = in[k*N + n]   (weights stored [K][N] in the reference)
__global__ void cvt_transpose_kernel(const float* __restrict__ in,
                                     bf16* __restrict__ out, int K, int N) {
  int i = blockIdx.x * 256 + threadIdx.x;
  if (i < K * N) {
    int n = i / K;
    int k = i - n * K;
    out[i] = (bf16)in[(size_t)k * N + n];
  }
}

// ---------------------------------------------------------------------------
// Shared GEMM mainloop.  WG = 256 thr / 8 waves; WG tile 128x128; K-step 32.
// The 128x32 B tile (shared by all 8 waves) is double-buffered in LDS and
// filled with async global->LDS B128 DMAs.  Per iteration:
//   wait_async + barrier  -> publish current buffer
//   kick next k-step DMA + prefetch next A fragment (overlaps the math)
//   load ALL 8 B frags to registers (batched ds_load -> partial dscnt waits)
//   8 chained WMMAs
// ---------------------------------------------------------------------------
static __device__ __forceinline__ void gemm_mainloop(
    const bf16* __restrict__ aRow, const bf16* __restrict__ WT, int n0,
    int tid, int row, int kh, bf16* Bsh, v8f acc[8]) {
  // This thread's two 16B staging chunks (512 chunks = 128 rows x 4 quarters).
  const int rn0 = tid >> 2, rq0 = (tid & 3) * 8;
  const int rn1 = rn0 + 64;
  const bf16* s0 = WT + (size_t)(n0 + rn0) * EMB + rq0;
  const bf16* s1 = WT + (size_t)(n0 + rn1) * EMB + rq0;
  const uint32_t lbase = (uint32_t)(uintptr_t)Bsh;   // LDS byte offset
  const uint32_t l0 = lbase + (uint32_t)((rn0 * 32 + rq0) * 2);
  const uint32_t l1 = lbase + (uint32_t)((rn1 * 32 + rq0) * 2);

  // Prologue: stage k = 0 into buffer 0; preload first A fragment.
  async_b128_x2(l0, l1, s0, s1);
  v16bf a_cur = pack16(aRow + kh * 8, aRow + 16 + kh * 8);

#pragma unroll 1
  for (int i = 0; i < KSTEPS; ++i) {
    const int k = i * 32;
    wait_async0();        // this wave's DMA for buffer (i&1) has landed
    __syncthreads();      // all waves' DMA visible; all done reading buf^1
    const int knext = (i + 1 < KSTEPS) ? k + 32 : 0;
    if (i + 1 < KSTEPS) { // overlap next tile's DMA with this tile's math
      const uint32_t boff = ((i + 1) & 1) ? (uint32_t)(128 * 32 * 2) : 0u;
      async_b128_x2(l0 + boff, l1 + boff, s0 + knext, s1 + knext);
    }
    v16bf a_next = pack16(aRow + knext + kh * 8, aRow + knext + 16 + kh * 8);

    const bf16* bb = Bsh + (size_t)(i & 1) * (128 * 32);
    v16bf bfrag[8];
#pragma unroll
    for (int t = 0; t < 8; ++t) {
      const bf16* bp = bb + (t * 16 + row) * 32 + kh * 8;
      bfrag[t] = pack16(bp, bp + 16);
    }
#pragma unroll
    for (int t = 0; t < 8; ++t) acc[t] = wmma_bf16(a_cur, bfrag[t], acc[t]);
    a_cur = a_next;
  }
}

// ---------------------------------------------------------------------------
// QKV GEMM:  [8192 x 1024] (bf16) x WqkvT [3072][1024] (bf16)
// Epilogue adds fp32 bias and scatters into Q[b][h][s][d], K[b][h][s][d],
// Vt[b][h][d][s]  (V transposed so the PV WMMA B-operand is contiguous).
// ---------------------------------------------------------------------------
__global__ void __launch_bounds__(256)
qkv_gemm_kernel(const bf16* __restrict__ Xb, const bf16* __restrict__ WT,
                const float* __restrict__ bias,
                bf16* __restrict__ Qh, bf16* __restrict__ Kh,
                bf16* __restrict__ Vt) {
  __shared__ bf16 Bsh[2][128 * 32];     // 16 KB double-buffered B tile
  const int lane = threadIdx.x & 31;
  const int wave = threadIdx.x >> 5;
  const int row  = lane & 15;
  const int kh   = lane >> 4;
  const int m0   = blockIdx.y * 128 + wave * 16;
  const int n0   = blockIdx.x * 128;

  v8f acc[8];
  const v8f vzero = {0.f, 0.f, 0.f, 0.f, 0.f, 0.f, 0.f, 0.f};
#pragma unroll
  for (int t = 0; t < 8; ++t) acc[t] = vzero;

  const bf16* aRow = Xb + (size_t)(m0 + row) * EMB;
  gemm_mainloop(aRow, WT, n0, threadIdx.x, row, kh, &Bsh[0][0], acc);

#pragma unroll
  for (int t = 0; t < 8; ++t) {
    const int n  = n0 + t * 16 + row;
    const float bv = bias[n];
    const int w  = n >> 10;           // 0=Q 1=K 2=V
    const int r  = n & 1023;
    const int hh = r >> 6;
    const int d  = r & 63;
#pragma unroll
    for (int j = 0; j < 8; ++j) {
      const int m  = m0 + j + 8 * kh;
      const int bb = m >> 11;         // batch (S = 2048)
      const int ss = m & 2047;        // seq
      const bf16 v = (bf16)(acc[t][j] + bv);
      const size_t bh = (size_t)(bb * HEADS + hh);
      if (w == 0)      Qh[(bh * SEQ + ss) * HD + d] = v;
      else if (w == 1) Kh[(bh * SEQ + ss) * HD + d] = v;
      else             Vt[(bh * HD + d) * SEQ + ss] = v;
    }
  }
}

// ---------------------------------------------------------------------------
// Flash attention.  grid = (B*H, S/64), block = 128 (4 waves).
// All 4 waves share each 32-key block, so K (32x64) and V (64x32) tiles are
// double-buffered in LDS and filled with async global->LDS DMAs (8 KB/step,
// 4 async B128 per thread), one wait+barrier per key block, next block's DMA
// overlapping this block's 8 WMMAs.  Each wave owns 16 query rows.
//   scores : 2 tiles x 2 WMMAs (contraction over head dim 64)
//   PV     : 4 WMMAs (contraction over the 32 keys, dims in 4 chunks of 16)
// P is re-striped C-layout -> A-layout through a private 1KB LDS tile.
// ---------------------------------------------------------------------------
__global__ void __launch_bounds__(128)
flash_attn_kernel(const bf16* __restrict__ Qh, const bf16* __restrict__ Kh,
                  const bf16* __restrict__ Vt, bf16* __restrict__ O) {
  __shared__ bf16 Ksh[2][32 * 64];      // 8 KB
  __shared__ bf16 Vsh[2][64 * 32];      // 8 KB
  __shared__ bf16 Psh[4][16 * 32];      // 4 KB

  const int tid  = threadIdx.x;
  const int lane = tid & 31;
  const int wave = tid >> 5;
  const int row  = lane & 15;
  const int kh   = lane >> 4;
  const int bh   = blockIdx.x;         // b*16 + h
  const int bb   = bh >> 4;
  const int hh   = bh & 15;
  const int q0   = blockIdx.y * 64 + wave * 16;

  const bf16* Qbase = Qh + (size_t)bh * SEQ * HD;
  const bf16* Kbase = Kh + (size_t)bh * SEQ * HD;
  const bf16* Vbase = Vt + (size_t)bh * HD * SEQ;

  // Per-thread staging chunks.
  // K tile: 32 rows(key) x 64 dims -> 256 B128 chunks: key = c>>3, q = c&7.
  // V tile: 64 rows(dim) x 32 keys -> 256 B128 chunks: dim = c>>2, q = c&3.
  const int keyq = tid >> 3, kq8 = (tid & 7) * 8;
  const int dimq = tid >> 2, vq8 = (tid & 3) * 8;
  const bf16* sK0 = Kbase + (size_t)keyq * HD + kq8;      // + kb*HD at issue
  const bf16* sK1 = sK0 + 16 * HD;
  const bf16* sV0 = Vbase + (size_t)dimq * SEQ + vq8;     // + kb at issue
  const bf16* sV1 = sV0 + 32 * SEQ;
  const uint32_t lK =
      (uint32_t)(uintptr_t)&Ksh[0][0] + (uint32_t)((keyq * 64 + kq8) * 2);
  const uint32_t lV =
      (uint32_t)(uintptr_t)&Vsh[0][0] + (uint32_t)((dimq * 32 + vq8) * 2);

  auto stage_kv = [&](int buf, int kb2) {
    const uint32_t bo = (uint32_t)buf * 4096u;
    async_b128_x2(lK + bo, lK + bo + 2048u,
                  sK0 + (size_t)kb2 * HD, sK1 + (size_t)kb2 * HD);
    async_b128_x2(lV + bo, lV + bo + 2048u, sV0 + kb2, sV1 + kb2);
  };

  // Q fragments: rows q0..q0+15, dims 0..63 -> two 16x32 A-frags
  v16bf aQ[2];
#pragma unroll
  for (int t = 0; t < 2; ++t) {
    const bf16* p = Qbase + (size_t)(q0 + row) * HD + t * 32;
    aQ[t] = pack16(p + kh * 8, p + 16 + kh * 8);
  }

  const v8f vzero = {0.f, 0.f, 0.f, 0.f, 0.f, 0.f, 0.f, 0.f};
  v8f o_acc[4];
#pragma unroll
  for (int t = 0; t < 4; ++t) o_acc[t] = vzero;
  float m_st[8], l_st[8];
#pragma unroll
  for (int j = 0; j < 8; ++j) { m_st[j] = -1e30f; l_st[j] = 0.f; }

  stage_kv(0, 0);       // prologue DMA

#pragma unroll 1
  for (int it = 0; it < SEQ / 32; ++it) {
    const int kb = it * 32;
    wait_async0();      // this wave's K/V DMA for buffer (it&1) has landed
    __syncthreads();    // publish; all waves done reading buf^1
    if (it + 1 < SEQ / 32) stage_kv((it + 1) & 1, kb + 32);

    const bf16* Kt = &Ksh[it & 1][0];
    const bf16* Vt_ = &Vsh[it & 1][0];

    // ---- scores: S = (Q*K^T) * 0.125, two 16x16 tiles over 32 keys ----
    v16bf bK[2][2];
#pragma unroll
    for (int u = 0; u < 2; ++u)
#pragma unroll
      for (int t = 0; t < 2; ++t) {
        const bf16* p = Kt + (u * 16 + row) * 64 + t * 32 + kh * 8;
        bK[u][t] = pack16(p, p + 16);
      }
    v8f s[2];
#pragma unroll
    for (int u = 0; u < 2; ++u) {
      v8f c = vzero;
      c = wmma_bf16(aQ[0], bK[u][0], c);
      c = wmma_bf16(aQ[1], bK[u][1], c);
      s[u] = c * 0.125f;
    }

    // ---- online softmax: row max / exp / row sum (16-lane reductions) ----
    float mnew[8], alpha[8];
#pragma unroll
    for (int j = 0; j < 8; ++j) {
      float mx = fmaxf(s[0][j], s[1][j]);
      mx = fmaxf(mx, __shfl_xor(mx, 1, 32));
      mx = fmaxf(mx, __shfl_xor(mx, 2, 32));
      mx = fmaxf(mx, __shfl_xor(mx, 4, 32));
      mx = fmaxf(mx, __shfl_xor(mx, 8, 32));
      mnew[j]  = fmaxf(m_st[j], mx);
      alpha[j] = __expf(m_st[j] - mnew[j]);
      m_st[j]  = mnew[j];
    }
#pragma unroll
    for (int u = 0; u < 2; ++u)
#pragma unroll
      for (int j = 0; j < 8; ++j) s[u][j] = __expf(s[u][j] - mnew[j]);
#pragma unroll
    for (int j = 0; j < 8; ++j) {
      float rs = s[0][j] + s[1][j];
      rs += __shfl_xor(rs, 1, 32);
      rs += __shfl_xor(rs, 2, 32);
      rs += __shfl_xor(rs, 4, 32);
      rs += __shfl_xor(rs, 8, 32);
      l_st[j] = l_st[j] * alpha[j] + rs;
    }
#pragma unroll
    for (int t = 0; t < 4; ++t)
#pragma unroll
      for (int j = 0; j < 8; ++j) o_acc[t][j] *= alpha[j];

    // ---- P: C-layout -> A-layout via per-wave LDS (same-wave DS order) ----
    bf16* lp = Psh[wave];
#pragma unroll
    for (int u = 0; u < 2; ++u)
#pragma unroll
      for (int j = 0; j < 8; ++j)
        lp[(j + 8 * kh) * 32 + u * 16 + row] = (bf16)s[u][j];

    const bf16* pr = lp + row * 32;
    v16bf aP = pack16(pr + kh * 8, pr + 16 + kh * 8);

    // ---- PV: contraction over 32 keys, dims in 4 chunks of 16 ----
    v16bf bV[4];
#pragma unroll
    for (int t = 0; t < 4; ++t) {
      const bf16* p = Vt_ + (t * 16 + row) * 32 + kh * 8;
      bV[t] = pack16(p, p + 16);
    }
#pragma unroll
    for (int t = 0; t < 4; ++t) o_acc[t] = wmma_bf16(aP, bV[t], o_acc[t]);
  }

  // ---- epilogue: O[b][s][h*64+d] = o_acc / l ----
#pragma unroll
  for (int j = 0; j < 8; ++j) {
    const float inv = 1.0f / l_st[j];
    const int ss = q0 + j + 8 * kh;
#pragma unroll
    for (int t = 0; t < 4; ++t) {
      const int d = t * 16 + row;
      O[((size_t)(bb * SEQ + ss)) * EMB + hh * HD + d] =
          (bf16)(o_acc[t][j] * inv);
    }
  }
}

// ---------------------------------------------------------------------------
// Output projection: [8192 x 1024] (bf16) x WprojT [1024][1024] (bf16)
// + fp32 bias -> fp32 output.  Same async/LDS-staged mainloop.
// ---------------------------------------------------------------------------
__global__ void __launch_bounds__(256)
proj_gemm_kernel(const bf16* __restrict__ Ob, const bf16* __restrict__ WT,
                 const float* __restrict__ bias, float* __restrict__ out) {
  __shared__ bf16 Bsh[2][128 * 32];
  const int lane = threadIdx.x & 31;
  const int wave = threadIdx.x >> 5;
  const int row  = lane & 15;
  const int kh   = lane >> 4;
  const int m0   = blockIdx.y * 128 + wave * 16;
  const int n0   = blockIdx.x * 128;

  v8f acc[8];
  const v8f vzero = {0.f, 0.f, 0.f, 0.f, 0.f, 0.f, 0.f, 0.f};
#pragma unroll
  for (int t = 0; t < 8; ++t) acc[t] = vzero;

  const bf16* aRow = Ob + (size_t)(m0 + row) * EMB;
  gemm_mainloop(aRow, WT, n0, threadIdx.x, row, kh, &Bsh[0][0], acc);

#pragma unroll
  for (int t = 0; t < 8; ++t) {
    const int n = n0 + t * 16 + row;
    const float bv = bias[n];
#pragma unroll
    for (int j = 0; j < 8; ++j) {
      const int m = m0 + j + 8 * kh;
      out[(size_t)m * EMB + n] = acc[t][j] + bv;
    }
  }
}

// ---------------------------------------------------------------------------
// Launcher.  Workspace layout (bf16 staging, all L2-resident):
//   Xb 16MB | WqkvT 6MB | WprojT 2MB | Q 16MB | K 16MB | Vt 16MB | O 16MB
// ---------------------------------------------------------------------------
extern "C" void kernel_launch(void* const* d_in, const int* in_sizes, int n_in,
                              void* d_out, int out_size, void* d_ws,
                              size_t ws_size, hipStream_t stream) {
  const float* X     = (const float*)d_in[0];
  const float* Wqkv  = (const float*)d_in[1];
  const float* bqkv  = (const float*)d_in[2];
  const float* Wproj = (const float*)d_in[3];
  const float* bproj = (const float*)d_in[4];
  float* out = (float*)d_out;

  char* ws = (char*)d_ws;
  bf16* Xb     = (bf16*)ws; ws += (size_t)ROWS * EMB * 2;
  bf16* WqkvT  = (bf16*)ws; ws += (size_t)3 * EMB * EMB * 2;
  bf16* WprojT = (bf16*)ws; ws += (size_t)EMB * EMB * 2;
  bf16* Qh     = (bf16*)ws; ws += (size_t)ROWS * EMB * 2;
  bf16* Kh     = (bf16*)ws; ws += (size_t)ROWS * EMB * 2;
  bf16* Vt     = (bf16*)ws; ws += (size_t)ROWS * EMB * 2;
  bf16* Ob     = (bf16*)ws; ws += (size_t)ROWS * EMB * 2;

  const int nX = ROWS * EMB;
  cvt_bf16_kernel<<<(nX + 255) / 256, 256, 0, stream>>>(X, Xb, nX);

  const int nWq = EMB * 3 * EMB;
  cvt_transpose_kernel<<<(nWq + 255) / 256, 256, 0, stream>>>(Wqkv, WqkvT, EMB,
                                                              3 * EMB);
  const int nWp = EMB * EMB;
  cvt_transpose_kernel<<<(nWp + 255) / 256, 256, 0, stream>>>(Wproj, WprojT,
                                                              EMB, EMB);

  qkv_gemm_kernel<<<dim3((3 * EMB) / 128, ROWS / 128), 256, 0, stream>>>(
      Xb, WqkvT, bqkv, Qh, Kh, Vt);

  flash_attn_kernel<<<dim3(BATCH * HEADS, SEQ / 64), 128, 0, stream>>>(
      Qh, Kh, Vt, Ob);

  proj_gemm_kernel<<<dim3(EMB / 128, ROWS / 128), 256, 0, stream>>>(
      Ob, WprojT, bproj, out);
}
